// BPTT_LSTM_67156108640262
// MI455X (gfx1250) — matrix operand order repaired and checked by reading, not compiled, not run on hardware
//
#include <hip/hip_runtime.h>
#include <stdint.h>

// ---- problem constants (T=512, B=64, I=H=1024) ----
#define T_STEPS 512
#define BATCH   64
#define HID     1024
#define KDIM    1024
#define GATES   4096            // 4*HID
#define MROWS   (T_STEPS*BATCH) // 32768
#define SCAN_BLOCKS 256         // persistent scan grid (4 M-tiles x 64 col-groups)
#define PAD 40                  // LDS row stride in bf16 elems (80B: 16B-aligned, conflict-free)

typedef __attribute__((ext_vector_type(16))) __bf16 v16bf;
typedef __attribute__((ext_vector_type(8)))  float  v8f;
typedef __attribute__((ext_vector_type(4)))  unsigned int u32x4;

union bfpack { u32x4 q[2]; v16bf v; };

__device__ __forceinline__ unsigned short f32_to_bf16_rne(float f) {
  union { float f; unsigned int u; } x; x.f = f;
  unsigned int u = x.u;
  unsigned int r = (u + 0x7FFFu + ((u >> 16) & 1u)) >> 16;
  return (unsigned short)r;
}

__device__ __forceinline__ float sigmoidf_(float x) {
  return 1.0f / (1.0f + __expf(-x));
}

// LDS byte offset of a __shared__ pointer: generic LDS address is
// {shared_aperture, offset32}; addr[31:0] is the wave-relative LDS offset.
__device__ __forceinline__ unsigned lds_off_of(const void* p) {
  return (unsigned)(uintptr_t)p;
}

// CDNA5 async memory->LDS copy (no VGPR transit, tracked by ASYNCcnt).
__device__ __forceinline__ void async_copy_b128(unsigned lds_off,
                                                const void* gptr) {
  asm volatile("global_load_async_to_lds_b128 %0, %1, off"
               :: "v"(lds_off), "v"(gptr) : "memory");
}
__device__ __forceinline__ void wait_async0() {
  asm volatile("s_wait_asynccnt 0x0" ::: "memory");
}

// ---- fp32 -> bf16 conversion, 4 elements/thread ----
__global__ void __launch_bounds__(256)
cvt_f32_bf16_x4(const float4* __restrict__ src, ushort4* __restrict__ dst, int n4) {
  int i = blockIdx.x * blockDim.x + threadIdx.x;
  if (i < n4) {
    float4 f = src[i];
    ushort4 o;
    o.x = f32_to_bf16_rne(f.x);
    o.y = f32_to_bf16_rne(f.y);
    o.z = f32_to_bf16_rne(f.z);
    o.w = f32_to_bf16_rne(f.w);
    dst[i] = o;
  }
}

// ---- Phase 1: xg[32768,4096] = Xbf * Wih_bf^T + (b_ih+b_hh)
// Block = 128x64 macro-tile; async global->LDS staging, statically
// unrolled 2-stage pipeline; 8 waves x 4 WMMA tiles (2Mx2N) via ds_load_b128.
__global__ void __launch_bounds__(256)
gemm_xgates(const unsigned short* __restrict__ Xb,
            const unsigned short* __restrict__ Wb,
            const float* __restrict__ bias_ih,
            const float* __restrict__ bias_hh,
            float* __restrict__ xg) {
  __shared__ __align__(16) unsigned short As[2][128 * PAD];
  __shared__ __align__(16) unsigned short Bs[2][64 * PAD];

  const int tid  = threadIdx.x;
  const int wave = tid >> 5;
  const int lane = tid & 31;
  const int bm   = blockIdx.x >> 6;   // 0..255
  const int bn   = blockIdx.x & 63;   // 0..63
  const int gm0  = bm * 128;
  const int gn0  = bn * 64;

  // staging map: A = 128 rows x 32 K (512 x 16B chunks, 2/thread), B = 64 rows x 32 K
  const int arow = tid >> 1;            // 0..127
  const int aseg = (tid & 1) * 2;       // {0,2} -> elements {0..15, 16..31}
  const int brow = tid >> 2;            // 0..63
  const int bseg = tid & 3;             // 16B chunk

  const unsigned short* Ag = Xb + (size_t)(gm0 + arow) * KDIM + aseg * 8;
  const unsigned short* Bg = Wb + (size_t)(gn0 + brow) * KDIM + bseg * 8;

  const unsigned a_l0 = lds_off_of(&As[0][arow * PAD + aseg * 8]);
  const unsigned a_l1 = lds_off_of(&As[0][arow * PAD + (aseg + 1) * 8]);
  const unsigned b_l0 = lds_off_of(&Bs[0][brow * PAD + bseg * 8]);
  const unsigned abuf = (unsigned)(128 * PAD * 2);  // bytes per A buffer
  const unsigned bbuf = (unsigned)(64 * PAD * 2);   // bytes per B buffer

  // wave tiling: 2 M-tiles x 2 N-tiles per wave
  const int mp = (wave >> 1) * 2;       // 0,2,4,6
  const int np = (wave & 1) * 2;        // 0,2

  const int rlow = lane & 15;
  const int aK   = (lane >> 4) * 8;     // A: hi-lanes start at K=8 (16B offset)
  const int bK   = (lane >> 4) * 8;     // B: hi-lanes start at K=8 (16B offset)

  v8f acc[2][2] = {{{}, {}}, {{}, {}}};

  auto stage = [&](unsigned bo, int k) {
    async_copy_b128(a_l0 + bo * abuf, Ag + k);
    async_copy_b128(a_l1 + bo * abuf, Ag + k + 8);
    async_copy_b128(b_l0 + bo * bbuf, Bg + k);
  };
  auto consume = [&](int pb) {
    bfpack a[2], b[2];
#pragma unroll
    for (int i = 0; i < 2; ++i) {
      const int ra = (mp + i) * 16 + rlow;
      a[i].q[0] = *(const u32x4*)&As[pb][ra * PAD + aK];
      a[i].q[1] = *(const u32x4*)&As[pb][ra * PAD + aK + 16];
      const int rb = (np + i) * 16 + rlow;
      b[i].q[0] = *(const u32x4*)&Bs[pb][rb * PAD + bK];
      b[i].q[1] = *(const u32x4*)&Bs[pb][rb * PAD + bK + 16];
    }
#pragma unroll
    for (int i = 0; i < 2; ++i)
#pragma unroll
      for (int j = 0; j < 2; ++j)
        acc[i][j] = __builtin_amdgcn_wmma_f32_16x16x32_bf16(
            false, a[i].v, false, b[j].v, (short)0, acc[i][j], false, false);
  };

  // 2-stage async pipeline, statically unrolled so buffer ids are constants
  stage(0u, 0);
  for (int kk = 0; kk < 32; kk += 2) {
    wait_async0();     // buf0 stores landed
    __syncthreads();   // all stores visible; prior buf1 reads complete
    stage(1u, (kk + 1) * 32);
    consume(0);
    wait_async0();     // buf1 stores landed
    __syncthreads();
    if (kk + 2 < 32) stage(0u, (kk + 2) * 32);
    consume(1);
  }

  // epilogue: add biases, store fp32
  const int colbase = gn0 + (lane & 15);
  const int rbv = (lane >> 4) * 8;
#pragma unroll
  for (int i = 0; i < 2; ++i) {
    const int m0 = gm0 + (mp + i) * 16 + rbv;
#pragma unroll
    for (int j = 0; j < 2; ++j) {
      const int col = colbase + (np + j) * 16;
      const float bs = bias_ih[col] + bias_hh[col];
      float* o = xg + (size_t)m0 * GATES + col;
#pragma unroll
      for (int v = 0; v < 8; ++v)
        o[(size_t)v * GATES] = acc[i][j][v] + bs;
    }
  }
}

// ---- Phase 2: persistent recurrent scan. One launch, 512 internal timesteps.
// Block b: mi = b>>6 (16 batch rows), n0 = (b&63)*16 (16 hidden cols).
// 8 waves = 4 gates x 2 K-halves; partials combined via LDS; device-wide
// software barrier (atomic + s_sleep spin + threadfence) between steps.
// Next timestep's xg tile is prefetched (global_prefetch_b8) during compute.
__global__ void __launch_bounds__(256)
lstm_scan(const float* __restrict__ xg,            // [512,64,4096]
          const unsigned short* __restrict__ Whh,  // [4096,1024] bf16
          unsigned short* __restrict__ hbuf,       // 2 x [64,1024] bf16 ping-pong
          float* __restrict__ c_state,             // [64,1024] fp32
          float* __restrict__ out,                 // [512,64,1024] fp32
          unsigned int* __restrict__ bar) {
  __shared__ float part[8][16][16];

  const int tid  = threadIdx.x;
  const int wave = tid >> 5;
  const int lane = tid & 31;
  const int mi   = blockIdx.x >> 6;      // 0..3
  const int n0   = (blockIdx.x & 63) * 16;
  const int g    = wave >> 1;            // gate 0..3 (i,f,g,o)
  const int kh   = wave & 1;             // K half
  const int kbase = kh * 512;

  const int row  = lane & 15;
  const int aK   = (lane >> 4) * 8;
  const int bK   = (lane >> 4) * 8;
  const int colL = lane & 15;
  const int rb   = (lane >> 4) * 8;

  const unsigned short* Brow =
      Whh + (size_t)(g * HID + n0 + row) * KDIM + kbase + bK;
  // per-lane xg tile base (used by kh==0 waves for C-init and prefetch)
  const size_t xg_tile = (size_t)(mi * 16 + rb) * GATES + (size_t)g * HID + n0 + colL;

  for (int t = 0; t < T_STEPS; ++t) {
    const unsigned short* h_in = hbuf + (size_t)(t & 1) * BATCH * HID;
    unsigned short* h_nx       = hbuf + (size_t)((t + 1) & 1) * BATCH * HID;
    const float* xg_t          = xg + (size_t)t * BATCH * GATES;

    const unsigned short* Arow =
        h_in + (size_t)(mi * 16 + row) * HID + kbase + aK;

    // C init: K-half 0 starts from the input projection, half 1 from zero
    v8f c = {};
    if (kh == 0) {
#pragma unroll
      for (int v = 0; v < 8; ++v)
        c[v] = xg_t[xg_tile + (size_t)v * GATES];
    }
#pragma unroll
    for (int k = 0; k < 512; k += 32) {
      bfpack a, b;
      a.q[0] = *(const u32x4*)(Arow + k);
      a.q[1] = *(const u32x4*)(Arow + k + 16);
      b.q[0] = *(const u32x4*)(Brow + k);
      b.q[1] = *(const u32x4*)(Brow + k + 16);
      c = __builtin_amdgcn_wmma_f32_16x16x32_bf16(false, a.v, false, b.v,
                                                  (short)0, c, false, false);
    }

    // hide next step's xg HBM latency behind this step's tail + barrier
    if (kh == 0 && t + 1 < T_STEPS) {
      const float* xg_n = xg + (size_t)(t + 1) * BATCH * GATES + xg_tile;
#pragma unroll
      for (int v = 0; v < 8; ++v)
        __builtin_prefetch(xg_n + (size_t)v * GATES, 0, 1);
    }

#pragma unroll
    for (int v = 0; v < 8; ++v) part[wave][rb + v][colL] = c[v];
    __syncthreads();

    // elementwise LSTM cell update: 256 threads -> 16x16 slice
    {
      const int r = tid >> 4, cc = tid & 15;
      const float ig = sigmoidf_(part[0][r][cc] + part[1][r][cc]);
      const float fg = sigmoidf_(part[2][r][cc] + part[3][r][cc]);
      const float gg = tanhf   (part[4][r][cc] + part[5][r][cc]);
      const float og = sigmoidf_(part[6][r][cc] + part[7][r][cc]);
      const size_t idx = (size_t)(mi * 16 + r) * HID + n0 + cc;
      const float cy = fg * c_state[idx] + ig * gg;
      const float hy = og * tanhf(cy);
      c_state[idx] = cy;
      out[(size_t)t * BATCH * HID + idx] = hy;
      h_nx[idx] = f32_to_bf16_rne(hy);
    }

    // ---- device-wide barrier between timesteps ----
    __threadfence();     // release h_nx / c_state to device scope
    __syncthreads();     // whole block done (also guards 'part' reuse)
    if (tid == 0) {
      __hip_atomic_fetch_add(bar, 1u, __ATOMIC_RELEASE, __HIP_MEMORY_SCOPE_AGENT);
      const unsigned int target = (unsigned int)(t + 1) * (unsigned int)SCAN_BLOCKS;
      while (__hip_atomic_load(bar, __ATOMIC_ACQUIRE, __HIP_MEMORY_SCOPE_AGENT) < target)
        __builtin_amdgcn_s_sleep(1);
    }
    __syncthreads();
    __threadfence();     // acquire: invalidate WGP$ before reading new h
  }
}

extern "C" void kernel_launch(void* const* d_in, const int* in_sizes, int n_in,
                              void* d_out, int out_size, void* d_ws, size_t ws_size,
                              hipStream_t stream) {
  const float* input     = (const float*)d_in[0]; // [512,64,1024]
  const float* initial_h = (const float*)d_in[1]; // [64,1024]
  const float* initial_c = (const float*)d_in[2]; // [64,1024]
  const float* weight_ih = (const float*)d_in[3]; // [4096,1024]
  const float* weight_hh = (const float*)d_in[4]; // [4096,1024]
  const float* bias_ih   = (const float*)d_in[5]; // [4096]
  const float* bias_hh   = (const float*)d_in[6]; // [4096]
  float* out = (float*)d_out; // [512,64,1024] h outputs then [64,1024] cT

  // ---- workspace layout ----
  char* ws = (char*)d_ws;
  size_t off = 0;
  float* xg = (float*)(ws + off);            off += (size_t)MROWS * GATES * sizeof(float); // 512 MB
  unsigned short* Xb    = (unsigned short*)(ws + off); off += (size_t)MROWS * KDIM * 2;    // 64 MB
  unsigned short* Wih_b = (unsigned short*)(ws + off); off += (size_t)GATES * KDIM * 2;    // 8 MB
  unsigned short* Whh_b = (unsigned short*)(ws + off); off += (size_t)GATES * KDIM * 2;    // 8 MB
  unsigned short* hbuf  = (unsigned short*)(ws + off); off += (size_t)2 * BATCH * HID * 2; // 256 KB
  float* c_state = (float*)(ws + off);       off += (size_t)BATCH * HID * sizeof(float);   // 256 KB
  unsigned int* bar = (unsigned int*)(ws + off); off += 256;                               // barrier ctr

  // ---- fp32 -> bf16 conversions ----
  {
    int n4 = (MROWS * KDIM) / 4;
    cvt_f32_bf16_x4<<<n4 / 256, 256, 0, stream>>>((const float4*)input, (ushort4*)Xb, n4);
  }
  {
    int n4 = (GATES * KDIM) / 4;
    cvt_f32_bf16_x4<<<n4 / 256, 256, 0, stream>>>((const float4*)weight_ih, (ushort4*)Wih_b, n4);
    cvt_f32_bf16_x4<<<n4 / 256, 256, 0, stream>>>((const float4*)weight_hh, (ushort4*)Whh_b, n4);
  }
  {
    int n4 = (BATCH * HID) / 4;
    cvt_f32_bf16_x4<<<n4 / 256, 256, 0, stream>>>((const float4*)initial_h, (ushort4*)hbuf, n4);
  }
  hipMemcpyAsync(c_state, initial_c, (size_t)BATCH * HID * sizeof(float),
                 hipMemcpyDeviceToDevice, stream);
  hipMemsetAsync(bar, 0, sizeof(unsigned int), stream);

  // ---- Phase 1: big GEMM for input projections (+biases) ----
  gemm_xgates<<<(MROWS / 128) * (GATES / 64), 256, 0, stream>>>(
      Xb, Wih_b, bias_ih, bias_hh, xg);

  // ---- Phase 2: single persistent kernel runs all 512 recurrent steps ----
  lstm_scan<<<SCAN_BLOCKS, 256, 0, stream>>>(xg, Whh_b, hbuf, c_state, out, bar);

  // ---- final cell state to output tail ----
  hipMemcpyAsync(out + (size_t)T_STEPS * BATCH * HID, c_state,
                 (size_t)BATCH * HID * sizeof(float),
                 hipMemcpyDeviceToDevice, stream);
}
